// BasicBlock_73873437491837
// MI455X (gfx1250) — matrix operand order repaired and checked by reading, not compiled, not run on hardware
//
#include <hip/hip_runtime.h>
#include <hip/hip_bf16.h>
#include <math.h>

#define BB 64
#define PW 33
#define PP 1089
#define MPHI 272
#define AP 39
#define APSZ 1521    // 39*39
#define NP1 9
#define NPP 81
#define KK 7
#define QK 784       // 16*49
#define QKPAD 800
#define SPAD 1120    // 1089 padded to mult of 32
#define PTB 8        // pixel tiles per conv block

typedef __attribute__((ext_vector_type(16))) __bf16 v16bf;
typedef __attribute__((ext_vector_type(8)))  __bf16 v8bf;
typedef __attribute__((ext_vector_type(8)))  float  v8f;

static __device__ __forceinline__ __bf16 tobf(float f) { return (__bf16)f; }

static __device__ __forceinline__ v8f wmma_bf16(v16bf a, v16bf b, v8f c) {
  return __builtin_amdgcn_wmma_f32_16x16x32_bf16(false, a, false, b, (short)0, c,
                                                 false, false);
}
static __device__ __forceinline__ v16bf cat8(v8bf lo, v8bf hi) {
  return __builtin_shufflevector(lo, hi, 0, 1, 2, 3, 4, 5, 6, 7,
                                 8, 9, 10, 11, 12, 13, 14, 15);
}
// A fragment: e[0..15] = K k0+2*half8+e (two 16B-aligned loads)
static __device__ __forceinline__ v16bf loadA(const __bf16* row, int k0, int half8) {
  v8bf lo = *(const v8bf*)(row + k0 + 2 * half8);
  v8bf hi = *(const v8bf*)(row + k0 + 2 * half8 + 8);
  return cat8(lo, hi);
}
// B fragment: e[0..15] = K k0+half16+e (two 16B-aligned loads)
static __device__ __forceinline__ v16bf loadB(const __bf16* col, int k0, int half16) {
  v8bf lo = *(const v8bf*)(col + k0 + half16);
  v8bf hi = *(const v8bf*)(col + k0 + half16 + 8);
  return cat8(lo, hi);
}

// ---------------------------------------------------------------------------
// Generic 3x3 same-padding conv on [B, Cin, 33, 33] via bf16 WMMA implicit GEMM.
// Compile-time shapes/flags; software-pipelined k-loop (prefetch next fragments
// before the WMMA on the current ones).
// ---------------------------------------------------------------------------
template <int CIN, int COUT, bool HAS_BIAS, bool HAS_RES, bool RELU>
__global__ void conv3x3_wmma_kernel(const float* __restrict__ in,
                                    const float* __restrict__ w,
                                    const float* __restrict__ bias,
                                    const float* __restrict__ residual,
                                    float* __restrict__ out) {
  constexpr int Kreal   = CIN * 9;
  constexpr int Kpad    = (Kreal + 31) & ~31;
  constexpr int CoutPad = (COUT + 15) & ~15;
  extern __shared__ char smem[];
  __bf16* lds_w = (__bf16*)smem;               // [CoutPad][Kpad]
  __bf16* lds_b = lds_w + CoutPad * Kpad;      // [PTB*16][Kpad]
  const int b = blockIdx.y;
  const int tid = threadIdx.x;
  const int pixBase = blockIdx.x * PTB * 16;

  for (int idx = tid; idx < CoutPad * Kpad; idx += 256) {
    int co = idx / Kpad, k = idx % Kpad;
    lds_w[idx] = tobf((co < COUT && k < Kreal) ? w[co * Kreal + k] : 0.f);
  }
  for (int idx = tid; idx < PTB * 16 * Kpad; idx += 256) {
    int col = idx / Kpad, k = idx % Kpad;
    int pix = pixBase + col;
    float val = 0.f;
    if (pix < PP && k < Kreal) {
      int ci = k / 9, rr = k - ci * 9;
      int dy = rr / 3, dx = rr - dy * 3;
      int y = pix / PW + dy - 1, x = pix % PW + dx - 1;
      if (y >= 0 && y < PW && x >= 0 && x < PW)
        val = in[((size_t)b * CIN + ci) * PP + y * PW + x];
    }
    lds_b[idx] = tobf(val);
  }
  __syncthreads();

  const int lane = tid & 31;
  const int wave = tid >> 5;
  const int half8  = (lane & 16) ? 8  : 0;
  const int half16 = (lane & 16) ? 16 : 0;
  const int ln = lane & 15;
  constexpr int coutTiles = CoutPad >> 4;
  constexpr int tiles = PTB * coutTiles;

  for (int t = wave; t < tiles; t += 8) {
    int pt = t % PTB, ct = t / PTB;
    int pixO = pixBase + pt * 16 + ln;
    if (pixBase + pt * 16 >= PP) continue;
    const __bf16* pa = lds_w + (ct * 16 + ln) * Kpad;
    const __bf16* pb = lds_b + (pt * 16 + ln) * Kpad;
    v8f acc = {};
    v16bf a0 = loadA(pa, 0, half8);
    v16bf b0 = loadB(pb, 0, half16);
#pragma unroll
    for (int k0 = 32; k0 < Kpad; k0 += 32) {
      v16bf a1 = loadA(pa, k0, half8);
      v16bf b1 = loadB(pb, k0, half16);
      acc = wmma_bf16(a0, b0, acc);
      a0 = a1; b0 = b1;
    }
    acc = wmma_bf16(a0, b0, acc);
#pragma unroll
    for (int r = 0; r < 8; r++) {
      int co = ct * 16 + r + half8;
      if (co < COUT && pixO < PP) {
        float val = acc[r];
        if (HAS_BIAS) val += bias[co];
        if (RELU) val = fmaxf(val, 0.f);
        size_t oidx = ((size_t)b * COUT + co) * PP + pixO;
        if (HAS_RES) val += residual[oidx];
        out[oidx] = val;
      }
    }
  }
}

// ---------------------------------------------------------------------------
// phix[b,m] = sum_i x[b,i] * Phi[m,i]   (64x1089 @ 1089x272, K pad 1120)
// ---------------------------------------------------------------------------
static __device__ __forceinline__ v16bf loadF32B(const float* p, int k0, int half16) {
  v16bf bf;
#pragma unroll
  for (int e = 0; e < 16; e++) bf[e] = tobf(p[k0 + half16 + e]);
  return bf;
}

__global__ void gemm_phi_kernel(const float* __restrict__ x,
                                const float* __restrict__ Phi,
                                float* __restrict__ out) {
  extern __shared__ char smem[];
  __bf16* lds_x = (__bf16*)smem;  // [64][1120]
  int tid = threadIdx.x;
  for (int idx = tid; idx < BB * SPAD; idx += 256) {
    int bb = idx / SPAD, i = idx % SPAD;
    lds_x[idx] = tobf(i < PP ? x[bb * PP + i] : 0.f);
  }
  __syncthreads();
  int lane = tid & 31, wave = tid >> 5;
  int half8 = (lane & 16) ? 8 : 0, half16 = (lane & 16) ? 16 : 0, ln = lane & 15;
  const int tot = 4 * 17;
  for (int t = wave; t < tot; t += 8) {
    int mt = t % 4, nt = t / 4;
    int rowA = mt * 16 + ln;     // batch row (<64)
    int colB = nt * 16 + ln;     // m (<272)
    const __bf16* pa = lds_x + rowA * SPAD;
    const float* pB = Phi + (size_t)colB * PP;
    v8f acc = {};
    v16bf a0 = loadA(pa, 0, half8);
    v16bf b0 = loadF32B(pB, 0, half16);
    for (int k0 = 32; k0 < 1088; k0 += 32) {   // k0+31 <= 1087 < 1089: no guard
      v16bf a1 = loadA(pa, k0, half8);
      v16bf b1 = loadF32B(pB, k0, half16);
      acc = wmma_bf16(a0, b0, acc);
      a0 = a1; b0 = b1;
    }
    acc = wmma_bf16(a0, b0, acc);
    {                                          // tail k0 = 1088 (guarded)
      v16bf bf;
#pragma unroll
      for (int e = 0; e < 16; e++) {
        int k = 1088 + half16 + e;
        bf[e] = tobf(k < PP ? pB[k] : 0.f);
      }
      acc = wmma_bf16(loadA(pa, 1088, half8), bf, acc);
    }
#pragma unroll
    for (int r = 0; r < 8; r++) {
      int bo = mt * 16 + r + half8;
      out[bo * MPHI + colB] = acc[r];
    }
  }
}

// ---------------------------------------------------------------------------
// out[b,i] = sum_m vec[b,m] * PhiT[i*272+m]   (64x272 @ 272x1089, K pad 288)
// ---------------------------------------------------------------------------
__global__ void gemm_phit_kernel(const float* __restrict__ vec,
                                 const float* __restrict__ PhiT,
                                 float* __restrict__ out) {
  extern __shared__ char smem[];
  __bf16* lds_v = (__bf16*)smem;  // [64][288]
  int tid = threadIdx.x;
  for (int idx = tid; idx < BB * 288; idx += 256) {
    int bb = idx / 288, m = idx % 288;
    lds_v[idx] = tobf(m < MPHI ? vec[bb * MPHI + m] : 0.f);
  }
  __syncthreads();
  int lane = tid & 31, wave = tid >> 5;
  int half8 = (lane & 16) ? 8 : 0, half16 = (lane & 16) ? 16 : 0, ln = lane & 15;
  const int tot = 4 * 69;
  for (int t = wave; t < tot; t += 8) {
    int mt = t % 4, nt = t / 4;
    int rowA = mt * 16 + ln;
    int colB = nt * 16 + ln; if (colB > PP - 1) colB = PP - 1;
    const __bf16* pa = lds_v + rowA * 288;
    const float* pB = PhiT + (size_t)colB * MPHI;
    v8f acc = {};
    v16bf a0 = loadA(pa, 0, half8);
    v16bf b0 = loadF32B(pB, 0, half16);
    for (int k0 = 32; k0 < 256; k0 += 32) {    // fully in-range
      v16bf a1 = loadA(pa, k0, half8);
      v16bf b1 = loadF32B(pB, k0, half16);
      acc = wmma_bf16(a0, b0, acc);
      a0 = a1; b0 = b1;
    }
    acc = wmma_bf16(a0, b0, acc);
    {                                           // tail k0 = 256 (guarded)
      v16bf bf;
#pragma unroll
      for (int e = 0; e < 16; e++) {
        int k = 256 + half16 + e;
        bf[e] = tobf(k < MPHI ? pB[k] : 0.f);
      }
      acc = wmma_bf16(loadA(pa, 256, half8), bf, acc);
    }
    int colO = nt * 16 + ln;
#pragma unroll
    for (int r = 0; r < 8; r++) {
      int bo = mt * 16 + r + half8;
      if (colO < PP) out[bo * PP + colO] = acc[r];
    }
  }
}

// ---------------------------------------------------------------------------
// scores[b,p,s]: per-image GEMM 81x784 @ 784x1089 (K pad 800).
// Q im2col panel staged once; per-ntile B panel built from the padded key field.
// ---------------------------------------------------------------------------
__global__ void attn_scores_kernel(const float* __restrict__ q16,
                                   const float* __restrict__ k16,
                                   float* __restrict__ scores) {
  extern __shared__ char smem[];
  __bf16* lds_qp = (__bf16*)smem;              // [96][800]
  __bf16* lds_kf = lds_qp + 96 * QKPAD;        // [16][1521]
  __bf16* lds_bp = lds_kf + 16 * APSZ;         // [16][800]
  int b = blockIdx.x, tid = threadIdx.x;
  for (int idx = tid; idx < 16 * APSZ; idx += 256) {
    int c = idx / APSZ, rem = idx % APSZ;
    int y = rem / AP - 3, x = rem % AP - 3;
    float kv = 0.f;
    if (y >= 0 && y < PW && x >= 0 && x < PW)
      kv = k16[((size_t)b * 16 + c) * PP + y * PW + x];
    lds_kf[idx] = tobf(kv);
  }
  for (int idx = tid; idx < 96 * QKPAD; idx += 256) {
    int p = idx / QKPAD, k = idx % QKPAD;
    float qv = 0.f;
    if (p < NPP && k < QK) {
      int c = k / 49, rr = k - c * 49;
      int dy = rr / KK, dx = rr - dy * KK;
      int y = (p / NP1) * 4 + dy - 3, x = (p % NP1) * 4 + dx - 3;
      if (y >= 0 && y < PW && x >= 0 && x < PW)
        qv = q16[((size_t)b * 16 + c) * PP + y * PW + x];
    }
    lds_qp[idx] = tobf(qv);
  }
  __syncthreads();
  int lane = tid & 31, wave = tid >> 5;
  int half8 = (lane & 16) ? 8 : 0, half16 = (lane & 16) ? 16 : 0, ln = lane & 15;
  for (int nt = 0; nt < 69; nt++) {
    for (int idx = tid; idx < 16 * QKPAD; idx += 256) {
      int n = idx / QKPAD, k = idx % QKPAD;
      int s = nt * 16 + n;
      __bf16 vv = (__bf16)0.f;
      if (s < PP && k < QK) {
        int c = k / 49, rr = k - c * 49;
        int dy = rr / KK, dx = rr - dy * KK;
        vv = lds_kf[c * APSZ + (s / PW + dy) * AP + (s % PW + dx)];
      }
      lds_bp[idx] = vv;
    }
    __syncthreads();
    if (wave < 6) {
      int mt = wave;
      const __bf16* pa = lds_qp + (mt * 16 + ln) * QKPAD;
      const __bf16* pb = lds_bp + ln * QKPAD;
      v8f acc = {};
      v16bf a0 = loadA(pa, 0, half8);
      v16bf b0 = loadB(pb, 0, half16);
      for (int k0 = 32; k0 < QKPAD; k0 += 32) {
        v16bf a1 = loadA(pa, k0, half8);
        v16bf b1 = loadB(pb, k0, half16);
        acc = wmma_bf16(a0, b0, acc);
        a0 = a1; b0 = b1;
      }
      acc = wmma_bf16(a0, b0, acc);
      int sO = nt * 16 + ln;
#pragma unroll
      for (int r = 0; r < 8; r++) {
        int po = mt * 16 + r + half8;
        if (po < NPP && sO < PP)
          scores[((size_t)b * NPP + po) * PP + sO] = acc[r];
      }
    }
    __syncthreads();
  }
}

// ---------------------------------------------------------------------------
// softmax over rows of [B*81, 1089] with scale 10 (in place)
// ---------------------------------------------------------------------------
__global__ void softmax_kernel(float* __restrict__ scores) {
  int row = blockIdx.x * 8 + (threadIdx.x >> 5);
  if (row >= BB * NPP) return;
  int lane = threadIdx.x & 31;
  float* base = scores + (size_t)row * PP;
  float m = -1e30f;
  for (int i = lane; i < PP; i += 32) m = fmaxf(m, base[i]);
  for (int off = 16; off; off >>= 1) m = fmaxf(m, __shfl_xor(m, off, 32));
  float sum = 0.f;
  for (int i = lane; i < PP; i += 32) {
    float e = __expf(10.0f * (base[i] - m));
    base[i] = e; sum += e;
  }
  for (int off = 16; off; off >>= 1) sum += __shfl_xor(sum, off, 32);
  float inv = 1.f / sum;
  for (int i = lane; i < PP; i += 32) base[i] *= inv;
}

// ---------------------------------------------------------------------------
// agg[b,p,kidx]: per-image GEMM 81x1089 @ 1089x784 (K pad 1120).
// attn panel staged once; per-ntile B panel built from the padded value field.
// ---------------------------------------------------------------------------
__global__ void attn_agg_kernel(const float* __restrict__ attn,
                                const float* __restrict__ v16,
                                float* __restrict__ agg) {
  extern __shared__ char smem[];
  __bf16* lds_ap = (__bf16*)smem;              // [96][1120]
  __bf16* lds_vf = lds_ap + 96 * SPAD;         // [16][1521]
  __bf16* lds_bp = lds_vf + 16 * APSZ;         // [16][1120]
  int b = blockIdx.x, tid = threadIdx.x;
  for (int idx = tid; idx < 96 * SPAD; idx += 256) {
    int p = idx / SPAD, s = idx % SPAD;
    lds_ap[idx] = tobf((p < NPP && s < PP) ? attn[((size_t)b * NPP + p) * PP + s] : 0.f);
  }
  for (int idx = tid; idx < 16 * APSZ; idx += 256) {
    int c = idx / APSZ, rem = idx % APSZ;
    int y = rem / AP - 3, x = rem % AP - 3;
    float vv = 0.f;
    if (y >= 0 && y < PW && x >= 0 && x < PW)
      vv = v16[((size_t)b * 16 + c) * PP + y * PW + x];
    lds_vf[idx] = tobf(vv);
  }
  __syncthreads();
  int lane = tid & 31, wave = tid >> 5;
  int half8 = (lane & 16) ? 8 : 0, half16 = (lane & 16) ? 16 : 0, ln = lane & 15;
  for (int nt = 0; nt < 49; nt++) {
    for (int idx = tid; idx < 16 * SPAD; idx += 256) {
      int n = idx / SPAD, s = idx % SPAD;
      int kidx = nt * 16 + n;
      int c = kidx / 49, rr = kidx - c * 49;
      int dy = rr / KK, dx = rr - dy * KK;
      __bf16 vv = (__bf16)0.f;
      if (s < PP)
        vv = lds_vf[c * APSZ + (s / PW + dy) * AP + (s % PW + dx)];
      lds_bp[idx] = vv;
    }
    __syncthreads();
    if (wave < 6) {
      int mt = wave;
      const __bf16* pa = lds_ap + (mt * 16 + ln) * SPAD;
      const __bf16* pb = lds_bp + ln * SPAD;
      v8f acc = {};
      v16bf a0 = loadA(pa, 0, half8);
      v16bf b0 = loadB(pb, 0, half16);
      for (int k0 = 32; k0 < SPAD; k0 += 32) {
        v16bf a1 = loadA(pa, k0, half8);
        v16bf b1 = loadB(pb, k0, half16);
        acc = wmma_bf16(a0, b0, acc);
        a0 = a1; b0 = b1;
      }
      acc = wmma_bf16(a0, b0, acc);
      int kidxO = nt * 16 + ln;
#pragma unroll
      for (int r = 0; r < 8; r++) {
        int po = mt * 16 + r + half8;
        if (po < NPP)
          agg[((size_t)b * NPP + po) * QK + kidxO] = acc[r];
      }
    }
    __syncthreads();
  }
}

// ---------------------------------------------------------------------------
// fold-back with overlap-count normalization + crop  -> y_att [B,16,33,33]
// ---------------------------------------------------------------------------
__global__ void fold_kernel(const float* __restrict__ agg, float* __restrict__ yatt) {
  int tid = blockIdx.x * 256 + threadIdx.x;
  if (tid >= BB * 16 * PP) return;
  int b = tid / (16 * PP), rem = tid % (16 * PP);
  int c = rem / PP, pix = rem % PP;
  int y = pix / PW, x = pix % PW;
  int Y = y + 3, X = x + 3;
  int dys[2], pys[2], ny = 0;
  int ry = Y & 3;
  for (int tt = 0; tt < 2; tt++) {
    int dy = ry + tt * 4;
    if (dy < KK) { int py = (Y - dy) >> 2; if (py >= 0 && py < NP1) { dys[ny] = dy; pys[ny] = py; ny++; } }
  }
  int dxs[2], pxs[2], nx = 0;
  int rx = X & 3;
  for (int tt = 0; tt < 2; tt++) {
    int dx = rx + tt * 4;
    if (dx < KK) { int px = (X - dx) >> 2; if (px >= 0 && px < NP1) { dxs[nx] = dx; pxs[nx] = px; nx++; } }
  }
  float sum = 0.f;
  for (int iy = 0; iy < ny; iy++)
    for (int ix = 0; ix < nx; ix++) {
      int p = pys[iy] * NP1 + pxs[ix];
      sum += agg[((size_t)b * NPP + p) * QK + c * 49 + dys[iy] * KK + dxs[ix]];
    }
  yatt[tid] = sum / (float)(ny * nx);
}

// ---------------------------------------------------------------------------
// small 1x1 projections + elementwise glue
// ---------------------------------------------------------------------------
__global__ void qkv1x1_kernel(const float* __restrict__ rr3,
                              const float* gw, const float* gb,
                              const float* tw, const float* tb,
                              const float* pw, const float* pb,
                              float* q, float* vv, float* kk) {
  int tid = blockIdx.x * 256 + threadIdx.x;
  if (tid >= BB * 16 * PP) return;
  int b = tid / (16 * PP), rem = tid % (16 * PP);
  int c = rem / PP, pix = rem % PP;
  float s1 = gb[c], s2 = tb[c], s3 = pb[c];
  const float* base = rr3 + (size_t)b * 64 * PP + pix;
  for (int ci = 0; ci < 64; ci++) {
    float v = base[ci * PP];
    s1 += gw[c * 64 + ci] * v;
    s2 += tw[c * 64 + ci] * v;
    s3 += pw[c * 64 + ci] * v;
  }
  q[tid] = s1; vv[tid] = s2; kk[tid] = s3;
}

__global__ void wproj_kernel(const float* __restrict__ yatt,
                             const float* Ww, const float* Wb,
                             float* __restrict__ rr3io) {
  int tid = blockIdx.x * 256 + threadIdx.x;
  if (tid >= BB * 64 * PP) return;
  int b = tid / (64 * PP), rem = tid % (64 * PP);
  int co = rem / PP, pix = rem % PP;
  float acc = Wb[co];
  const float* base = yatt + (size_t)b * 16 * PP + pix;
  for (int ci = 0; ci < 16; ci++) acc += Ww[co * 16 + ci] * base[ci * PP];
  rr3io[tid] += acc;   // y = W(y_att) + b + rr3, stored in place
}

__global__ void conv1ch_kernel(const float* __restrict__ in, const float* __restrict__ w9,
                               float* __restrict__ out, int relu_flag) {
  int tid = blockIdx.x * 256 + threadIdx.x;
  if (tid >= BB * PP) return;
  int b = tid / PP, pix = tid % PP;
  int y = pix / PW, x = pix % PW;
  float acc = 0.f;
  for (int dy = 0; dy < 3; dy++)
    for (int dx = 0; dx < 3; dx++) {
      int yy = y + dy - 1, xxp = x + dx - 1;
      if (yy >= 0 && yy < PW && xxp >= 0 && xxp < PW)
        acc += w9[dy * 3 + dx] * in[(size_t)b * PP + yy * PW + xxp];
    }
  if (relu_flag) acc = fmaxf(acc, 0.f);
  out[tid] = acc;
}

__global__ void e_duv_kernel(const float* du, const float* v, const float* phitphi,
                             const float* PhiTb, const float* beta_p, const float* ls2_p,
                             float* duv) {
  int i = blockIdx.x * 256 + threadIdx.x;
  if (i >= BB * PP) return;
  float beta = beta_p[0], ls2 = ls2_p[0];
  duv[i] = du[i] - v[i] / beta - ls2 * phitphi[i] + ls2 * PhiTb[i];
}

__global__ void e_aubru_kernel(const float* phix, const float* Phix, const float* ru,
                               const float* mu_p, float* aubru) {
  int i = blockIdx.x * 256 + threadIdx.x;
  if (i >= BB * MPHI) return;
  aubru[i] = mu_p[0] * (phix[i] - Phix[i]) - ru[i];
}

__global__ void e_duvw_kernel(const float* du, const float* v, const float* xpred,
                              const float* beta_p, float* duvw) {
  int i = blockIdx.x * 256 + threadIdx.x;
  if (i >= BB * PP) return;
  float beta = beta_p[0];
  duvw[i] = beta * du[i] - v[i] - beta * xpred[i];
}

__global__ void e_rr_kernel(const float* duvww, const float* r, const float* x,
                            const float* xx, const float* wa, const float* uu,
                            const float* theta_p, const float* ls_p,
                            const float* bn_g, const float* bn_b,
                            float* rr, float* bn1a) {
  int i = blockIdx.x * 256 + threadIdx.x;
  if (i >= BB * PP) return;
  float theta = theta_p[0], ls = ls_p[0];
  float d = duvww[i] - r[i] + theta * (x[i] - xx[i]) + wa[i];
  float uu2 = uu[i] - ls * d;
  float rrv = uu2 - r[i] / theta;
  rr[i] = rrv;
  bn1a[i] = bn_g[0] * rrv * rsqrtf(1.0f + 1e-5f) + bn_b[0];
}

__global__ void e_st_kernel(const float* xf, const float* thr_p, float* st) {
  int i = blockIdx.x * 256 + threadIdx.x;
  if (i >= BB * 32 * PP) return;
  float t = thr_p[0], v = xf[i];
  float m = fabsf(v) - t; m = m > 0.f ? m : 0.f;
  st[i] = (v >= 0.f) ? m : -m;
}

// ---------------------------------------------------------------------------
template <int CIN, int COUT, bool HAS_BIAS, bool HAS_RES, bool RELU>
static void launch_conv_t(const float* in, const float* w, const float* bias,
                          const float* res, float* o, hipStream_t stream) {
  constexpr int Kpad = (CIN * 9 + 31) & ~31;
  constexpr int CoutPad = (COUT + 15) & ~15;
  size_t shmem = ((size_t)CoutPad * Kpad + (size_t)PTB * 16 * Kpad) * 2;
  conv3x3_wmma_kernel<CIN, COUT, HAS_BIAS, HAS_RES, RELU>
      <<<dim3(9, BB), 256, shmem, stream>>>(in, w, bias, res, o);
}

extern "C" void kernel_launch(void* const* d_in, const int* in_sizes, int n_in,
                              void* d_out, int out_size, void* d_ws, size_t ws_size,
                              hipStream_t stream) {
  (void)in_sizes; (void)n_in; (void)out_size; (void)ws_size;
  const float* x      = (const float*)d_in[0];
  const float* PhiTb  = (const float*)d_in[1];
  const float* v      = (const float*)d_in[2];
  const float* r      = (const float*)d_in[3];
  const float* ru     = (const float*)d_in[4];
  const float* xx     = (const float*)d_in[5];
  const float* uu     = (const float*)d_in[6];
  const float* Phix   = (const float*)d_in[7];
  const float* PhiW   = (const float*)d_in[8];
  const float* PhiTW  = (const float*)d_in[9];
  const float* ls_p   = (const float*)d_in[10];
  const float* ls2_p  = (const float*)d_in[11];
  const float* thr_p  = (const float*)d_in[12];
  const float* beta_p = (const float*)d_in[13];
  const float* mu_p   = (const float*)d_in[14];
  const float* th_p   = (const float*)d_in[15];
  const float* conv1_w  = (const float*)d_in[16];
  const float* conv2_w  = (const float*)d_in[17];
  const float* convG1_w = (const float*)d_in[18];
  const float* convD_w  = (const float*)d_in[19];
  const float* conv1f_w = (const float*)d_in[20];
  const float* conv2f_w = (const float*)d_in[21];
  const float* conv1b_w = (const float*)d_in[22];
  const float* conv2b_w = (const float*)d_in[23];
  const float* convG_w  = (const float*)d_in[24];
  const float* convG2_w = (const float*)d_in[25];
  const float* head_w = (const float*)d_in[26];
  const float* head_b = (const float*)d_in[27];
  const float* tail_w = (const float*)d_in[28];
  const float* tail_b = (const float*)d_in[29];
  const float* rb1_w1 = (const float*)d_in[30];
  const float* rb1_b1 = (const float*)d_in[31];
  const float* rb1_w2 = (const float*)d_in[32];
  const float* rb1_b2 = (const float*)d_in[33];
  const float* rb2_w1 = (const float*)d_in[34];
  const float* rb2_b1 = (const float*)d_in[35];
  const float* rb2_w2 = (const float*)d_in[36];
  const float* rb2_b2 = (const float*)d_in[37];
  const float* g_w  = (const float*)d_in[38];
  const float* g_b  = (const float*)d_in[39];
  const float* t_w  = (const float*)d_in[40];
  const float* t_b  = (const float*)d_in[41];
  const float* p_w  = (const float*)d_in[42];
  const float* p_b  = (const float*)d_in[43];
  const float* W_w  = (const float*)d_in[44];
  const float* W_b  = (const float*)d_in[45];
  const float* bn_g = (const float*)d_in[46];
  const float* bn_b = (const float*)d_in[47];
  float* out = (float*)d_out;

  float* ws = (float*)d_ws;
  size_t off = 0;
  auto alloc = [&](size_t n) { float* p = ws + off; off += n; return p; };
  float* du      = alloc((size_t)BB * PP);
  float* phix    = alloc((size_t)BB * MPHI);
  float* aubru   = alloc((size_t)BB * MPHI);
  float* phitphi = alloc((size_t)BB * PP);
  float* wa      = alloc((size_t)BB * PP);
  float* duv     = alloc((size_t)BB * PP);
  float* xpred   = alloc((size_t)BB * PP);
  float* duvw    = alloc((size_t)BB * PP);
  float* duvww   = alloc((size_t)BB * PP);
  float* rr      = alloc((size_t)BB * PP);
  float* bn1a    = alloc((size_t)BB * PP);
  float* rrnl    = alloc((size_t)BB * PP);
  float* c32a    = alloc((size_t)BB * 32 * PP);
  float* c32b    = alloc((size_t)BB * 32 * PP);
  float* c32c    = alloc((size_t)BB * 32 * PP);
  float* rr3     = alloc((size_t)BB * 64 * PP);
  float* b1t     = alloc((size_t)BB * 16 * PP);
  float* b2t     = alloc((size_t)BB * 16 * PP);
  float* b3t     = alloc((size_t)BB * 16 * PP);
  float* scores  = alloc((size_t)BB * NPP * PP);
  float* agg     = alloc((size_t)BB * NPP * QK);
  float* yatt    = alloc((size_t)BB * 16 * PP);

  const int GPIX = (BB * PP + 255) / 256;

  // --- gradient-descent stage ---
  conv1ch_kernel<<<GPIX, 256, 0, stream>>>(x, conv1_w, du, 1);
  gemm_phi_kernel<<<1, 256, (size_t)BB * SPAD * 2, stream>>>(x, PhiW, phix);
  e_aubru_kernel<<<(BB * MPHI + 255) / 256, 256, 0, stream>>>(phix, Phix, ru, mu_p, aubru);
  gemm_phit_kernel<<<1, 256, (size_t)BB * 288 * 2, stream>>>(phix, PhiTW, phitphi);
  gemm_phit_kernel<<<1, 256, (size_t)BB * 288 * 2, stream>>>(aubru, PhiTW, wa);
  e_duv_kernel<<<GPIX, 256, 0, stream>>>(du, v, phitphi, PhiTb, beta_p, ls2_p, duv);

  // --- head / resblocks / tail ---
  launch_conv_t<1, 32, true, false, false>(duv,  head_w, head_b, nullptr, c32a, stream);
  launch_conv_t<32, 32, true, false, true >(c32a, rb1_w1, rb1_b1, nullptr, c32b, stream);
  launch_conv_t<32, 32, true, true,  false>(c32b, rb1_w2, rb1_b2, c32a,    c32c, stream);
  launch_conv_t<32, 32, true, false, true >(c32c, rb2_w1, rb2_b1, nullptr, c32b, stream);
  launch_conv_t<32, 32, true, true,  false>(c32b, rb2_w2, rb2_b2, c32c,    c32a, stream);
  launch_conv_t<32, 1,  true, true,  false>(c32a, tail_w, tail_b, duv,     xpred, stream);

  e_duvw_kernel<<<GPIX, 256, 0, stream>>>(du, v, xpred, beta_p, duvw);
  conv1ch_kernel<<<GPIX, 256, 0, stream>>>(duvw, conv2_w, duvww, 1);
  e_rr_kernel<<<GPIX, 256, 0, stream>>>(duvww, r, x, xx, wa, uu, th_p, ls_p,
                                        bn_g, bn_b, rr, bn1a);

  // --- non-local attention ---
  launch_conv_t<1, 64, false, false, true>(bn1a, convG1_w, nullptr, nullptr, rr3, stream);
  qkv1x1_kernel<<<(BB * 16 * PP + 255) / 256, 256, 0, stream>>>(
      rr3, g_w, g_b, t_w, t_b, p_w, p_b, b1t, b2t, b3t);
  attn_scores_kernel<<<BB, 256,
      ((size_t)96 * QKPAD + (size_t)16 * APSZ + (size_t)16 * QKPAD) * 2, stream>>>(
      b1t, b3t, scores);
  softmax_kernel<<<(BB * NPP + 7) / 8, 256, 0, stream>>>(scores);
  attn_agg_kernel<<<BB, 256,
      ((size_t)96 * SPAD + (size_t)16 * APSZ + (size_t)16 * SPAD) * 2, stream>>>(
      scores, b2t, agg);
  fold_kernel<<<(BB * 16 * PP + 255) / 256, 256, 0, stream>>>(agg, yatt);
  wproj_kernel<<<(BB * 64 * PP + 255) / 256, 256, 0, stream>>>(yatt, W_w, W_b, rr3);
  launch_conv_t<64, 1, false, true, false>(rr3, convG2_w, nullptr, rr, rrnl, stream);

  // --- ISTA soft-threshold stage ---
  launch_conv_t<1, 32, false, false, false>(rrnl, convD_w,  nullptr, nullptr, c32a, stream);
  launch_conv_t<32, 32, false, false, true >(c32a, conv1f_w, nullptr, nullptr, c32b, stream);
  launch_conv_t<32, 32, false, false, false>(c32b, conv2f_w, nullptr, nullptr, c32c, stream);
  e_st_kernel<<<(BB * 32 * PP + 255) / 256, 256, 0, stream>>>(c32c, thr_p, c32a);
  launch_conv_t<32, 32, false, false, true >(c32a, conv1b_w, nullptr, nullptr, c32b, stream);
  launch_conv_t<32, 32, false, false, false>(c32b, conv2b_w, nullptr, nullptr, c32c, stream);
  launch_conv_t<32, 1,  false, true,  false>(c32c, convG_w,  nullptr, rrnl,    out, stream);
}
